// GATv2_63668595196559
// MI455X (gfx1250) — compile-verified, matrix operand order checked by reading, not run on hardware
//
#include <hip/hip_runtime.h>
#include <math.h>

// GATv2: N nodes, E edges, F=128 in-features, H=8 heads, D=16 -> HD=128 out dims.
#define FDIM 128
#define HDIM 8
#define DDIM 16
#define HD   128

typedef __attribute__((ext_vector_type(16))) _Float16 v16h;
typedef __attribute__((ext_vector_type(8)))  _Float16 v8h;
typedef __attribute__((ext_vector_type(8)))  float    v8f;

__device__ __forceinline__ float mish_f(float x) {
    float sp = (x > 15.0f) ? x : log1pf(__expf(x));
    return x * tanhf(sp);
}

__device__ __forceinline__ void atomicMaxFloat(float* addr, float val) {
    unsigned int* ua = (unsigned int*)addr;
    unsigned int old = *ua;
    while (__uint_as_float(old) < val) {
        unsigned int assumed = old;
        old = atomicCAS(ua, assumed, __float_as_uint(val));
        if (old == assumed) break;
    }
}

// ---------------------------------------------------------------------------
// Kernel 0: initialize output + segment-softmax scratch (graph-replay safe).
// ---------------------------------------------------------------------------
__global__ __launch_bounds__(256) void init_kernel(float* __restrict__ out,
                                                   float* __restrict__ maxbuf,
                                                   float* __restrict__ sumbuf,
                                                   long nOut, long nSeg) {
    long i = (long)blockIdx.x * blockDim.x + threadIdx.x;
    if (i < nOut) out[i] = 0.0f;
    if (i < nSeg) {
        maxbuf[i] = -INFINITY;
        sumbuf[i] = 0.0f;
    }
}

// ---------------------------------------------------------------------------
// Kernel 1: Ps = nodes @ Ws + bs ; Pr = nodes @ Wr + br   (WMMA f16->f32)
// Block = 256 threads = 8 waves. Block computes a 128-row strip of both
// projections; wave w owns rows [w*16, w*16+16) x all 128 columns.
// K stepped by 32 with v_wmma_f32_16x16x32_f16. A and B tiles staged in LDS
// as f16, laid out [row/col][K] so each fragment is two ds_load_b128 per lane.
// ---------------------------------------------------------------------------
__global__ __launch_bounds__(256) void gemm_proj_kernel(
    const float* __restrict__ nodes,
    const float* __restrict__ Ws, const float* __restrict__ bs,
    const float* __restrict__ Wr, const float* __restrict__ br,
    float* __restrict__ Ps, float* __restrict__ Pr, int N)
{
    __shared__ _Float16 la [128 * 32];  // A tile: [local_row][k]   8 KB
    __shared__ _Float16 lbs[128 * 32];  // Ws tile: [col][k]        8 KB
    __shared__ _Float16 lbr[128 * 32];  // Wr tile: [col][k]        8 KB

    const int tid  = threadIdx.x;
    const int wave = tid >> 5;
    const int lane = tid & 31;
    const int g    = lane >> 4;   // half-wave group: K-offset select
    const int mr   = lane & 15;   // row (A) / col (B) within 16x16 tile
    const int blockRow = blockIdx.x * 128;

    v8f accS[8], accR[8];
#pragma unroll
    for (int c = 0; c < 8; ++c)
#pragma unroll
        for (int r = 0; r < 8; ++r) { accS[c][r] = 0.0f; accR[c][r] = 0.0f; }

    for (int kk0 = 0; kk0 < FDIM; kk0 += 32) {
        // ---- stage A: nodes[blockRow .. +127][kk0 .. +31] -> f16 ----
        {
            const int rowL = tid >> 1;            // 0..127
            const int half = (tid & 1) * 16;      // 0 or 16
            const int grow = blockRow + rowL;
            _Float16* dst = &la[rowL * 32 + half];
            if (grow < N) {
                const float4* src =
                    (const float4*)(nodes + (size_t)grow * FDIM + kk0 + half);
#pragma unroll
                for (int q = 0; q < 4; ++q) {
                    float4 v = src[q];
                    dst[q * 4 + 0] = (_Float16)v.x;
                    dst[q * 4 + 1] = (_Float16)v.y;
                    dst[q * 4 + 2] = (_Float16)v.z;
                    dst[q * 4 + 3] = (_Float16)v.w;
                }
            } else {
#pragma unroll
                for (int q = 0; q < 16; ++q) dst[q] = (_Float16)0.0f;
            }
        }
        // ---- stage B (both weights), transposed to [col][k] ----
        {
            const int kL      = tid >> 3;          // 0..31
            const int colBase = (tid & 7) * 16;    // 0..112
            const float4* srcS =
                (const float4*)(Ws + (size_t)(kk0 + kL) * HD + colBase);
            const float4* srcR =
                (const float4*)(Wr + (size_t)(kk0 + kL) * HD + colBase);
#pragma unroll
            for (int q = 0; q < 4; ++q) {
                float4 vs = srcS[q];
                float4 vr = srcR[q];
                lbs[(colBase + q * 4 + 0) * 32 + kL] = (_Float16)vs.x;
                lbs[(colBase + q * 4 + 1) * 32 + kL] = (_Float16)vs.y;
                lbs[(colBase + q * 4 + 2) * 32 + kL] = (_Float16)vs.z;
                lbs[(colBase + q * 4 + 3) * 32 + kL] = (_Float16)vs.w;
                lbr[(colBase + q * 4 + 0) * 32 + kL] = (_Float16)vr.x;
                lbr[(colBase + q * 4 + 1) * 32 + kL] = (_Float16)vr.y;
                lbr[(colBase + q * 4 + 2) * 32 + kL] = (_Float16)vr.z;
                lbr[(colBase + q * 4 + 3) * 32 + kL] = (_Float16)vr.w;
            }
        }
        __syncthreads();

        // ---- A fragment for this wave's 16-row strip ----
        const _Float16* ap = &la[(wave * 16 + mr) * 32];
        v8h alo = *(const v8h*)(ap + 8 * g);        // K = 8g .. 8g+7
        v8h ahi = *(const v8h*)(ap + 16 + 8 * g);   // K = 16+8g .. 23+8g
        v16h af;
#pragma unroll
        for (int i = 0; i < 8; ++i) { af[i] = alo[i]; af[i + 8] = ahi[i]; }

        // ---- 8 column tiles x 2 weight matrices ----
#pragma unroll
        for (int c = 0; c < 8; ++c) {
            const _Float16* bpS = &lbs[(c * 16 + mr) * 32];
            v8h blo = *(const v8h*)(bpS + 8 * g);
            v8h bhi = *(const v8h*)(bpS + 16 + 8 * g);
            v16h bf;
#pragma unroll
            for (int i = 0; i < 8; ++i) { bf[i] = blo[i]; bf[i + 8] = bhi[i]; }
            accS[c] = __builtin_amdgcn_wmma_f32_16x16x32_f16(
                false, af, false, bf, (short)0, accS[c], false, false);

            const _Float16* bpR = &lbr[(c * 16 + mr) * 32];
            blo = *(const v8h*)(bpR + 8 * g);
            bhi = *(const v8h*)(bpR + 16 + 8 * g);
#pragma unroll
            for (int i = 0; i < 8; ++i) { bf[i] = blo[i]; bf[i + 8] = bhi[i]; }
            accR[c] = __builtin_amdgcn_wmma_f32_16x16x32_f16(
                false, af, false, bf, (short)0, accR[c], false, false);
        }
        __syncthreads();
    }

    // ---- epilogue: bias + store. C/D layout: VGPR r -> M = r + 8*g ----
#pragma unroll
    for (int c = 0; c < 8; ++c) {
        const int col = c * 16 + mr;
        const float bsv = bs[col];
        const float brv = br[col];
#pragma unroll
        for (int r = 0; r < 8; ++r) {
            const int row = blockRow + wave * 16 + g * 8 + r;
            if (row < N) {
                Ps[(size_t)row * HD + col] = accS[c][r] + bsv;
                Pr[(size_t)row * HD + col] = accR[c][r] + brv;
            }
        }
    }
}

// ---------------------------------------------------------------------------
// Kernel 2: per-(edge,head) attention logit + segment max.
// ---------------------------------------------------------------------------
__global__ __launch_bounds__(256) void logits_kernel(
    const float* __restrict__ Ps, const float* __restrict__ Pr,
    const int* __restrict__ senders, const int* __restrict__ receivers,
    const float* __restrict__ attn_k, const float* __restrict__ attn_b,
    float* __restrict__ logits, float* __restrict__ maxbuf, long EH)
{
    long i = (long)blockIdx.x * blockDim.x + threadIdx.x;
    if (i >= EH) return;
    const int  h = (int)(i & (HDIM - 1));
    const long e = i >> 3;
    const int  s = senders[e];
    const int  r = receivers[e];
    const float4* ps = (const float4*)(Ps + (size_t)s * HD + h * DDIM);
    const float4* pr = (const float4*)(Pr + (size_t)r * HD + h * DDIM);
    const float4* ak = (const float4*)attn_k;
    float acc = attn_b[0];
#pragma unroll
    for (int q = 0; q < 4; ++q) {
        float4 a = ps[q];
        float4 b = pr[q];
        float4 k = ak[q];
        acc += mish_f(a.x + b.x) * k.x;
        acc += mish_f(a.y + b.y) * k.y;
        acc += mish_f(a.z + b.z) * k.z;
        acc += mish_f(a.w + b.w) * k.w;
    }
    logits[i] = acc;
    atomicMaxFloat(maxbuf + (size_t)r * HDIM + h, acc);
}

// ---------------------------------------------------------------------------
// Kernel 3: exp(logit - segmax) in place + segment sum.
// ---------------------------------------------------------------------------
__global__ __launch_bounds__(256) void expsum_kernel(
    const int* __restrict__ receivers,
    float* __restrict__ logits,
    const float* __restrict__ maxbuf, float* __restrict__ sumbuf, long EH)
{
    long i = (long)blockIdx.x * blockDim.x + threadIdx.x;
    if (i >= EH) return;
    const int  h = (int)(i & (HDIM - 1));
    const long e = i >> 3;
    const int  r = receivers[e];
    const float v = __expf(logits[i] - maxbuf[(size_t)r * HDIM + h]);
    logits[i] = v;
    atomicAdd(sumbuf + (size_t)r * HDIM + h, v);
}

// ---------------------------------------------------------------------------
// Kernel 4: weighted scatter of sender projections into receiver nodes.
// ---------------------------------------------------------------------------
__global__ __launch_bounds__(256) void scatter_kernel(
    const float* __restrict__ Ps,
    const int* __restrict__ senders, const int* __restrict__ receivers,
    const float* __restrict__ logits, const float* __restrict__ sumbuf,
    float* __restrict__ out, long EH)
{
    long i = (long)blockIdx.x * blockDim.x + threadIdx.x;
    if (i >= EH) return;
    const int  h = (int)(i & (HDIM - 1));
    const long e = i >> 3;
    const int  s = senders[e];
    const int  r = receivers[e];
    const float w = logits[i] / sumbuf[(size_t)r * HDIM + h];
    const float4* ps = (const float4*)(Ps + (size_t)s * HD + h * DDIM);
    float* op = out + (size_t)r * HD + h * DDIM;
#pragma unroll
    for (int q = 0; q < 4; ++q) {
        float4 a = ps[q];
        atomicAdd(op + q * 4 + 0, a.x * w);
        atomicAdd(op + q * 4 + 1, a.y * w);
        atomicAdd(op + q * 4 + 2, a.z * w);
        atomicAdd(op + q * 4 + 3, a.w * w);
    }
}

// ---------------------------------------------------------------------------
// Host launcher.
// Inputs: 0=nodes[N,128] 1=senders[E] 2=receivers[E] 3=Ws_k[128,8,16]
//         4=Ws_b[8,16] 5=Wr_k 6=Wr_b 7=attn_k[16,1] 8=attn_b[1]
// ---------------------------------------------------------------------------
extern "C" void kernel_launch(void* const* d_in, const int* in_sizes, int n_in,
                              void* d_out, int out_size, void* d_ws, size_t ws_size,
                              hipStream_t stream) {
    const float* nodes     = (const float*)d_in[0];
    const int*   senders   = (const int*)  d_in[1];
    const int*   receivers = (const int*)  d_in[2];
    const float* Ws        = (const float*)d_in[3];
    const float* bs        = (const float*)d_in[4];
    const float* Wr        = (const float*)d_in[5];
    const float* br        = (const float*)d_in[6];
    const float* attn_k    = (const float*)d_in[7];
    const float* attn_b    = (const float*)d_in[8];
    float*       out       = (float*)d_out;

    const int  N  = in_sizes[0] / FDIM;
    const long E  = (long)in_sizes[1];
    const long EH = E * HDIM;

    // Workspace carve-up (256B aligned): Ps, Pr, logits/exp (in place),
    // segment max, segment sum.
    auto al = [](size_t x) { return (x + 255) & ~(size_t)255; };
    char*  ws    = (char*)d_ws;
    size_t szP   = al((size_t)N * HD * sizeof(float));
    size_t szLg  = al((size_t)EH * sizeof(float));
    size_t szSeg = al((size_t)N * HDIM * sizeof(float));
    float* Ps     = (float*)(ws);
    float* Pr     = (float*)(ws + szP);
    float* logits = (float*)(ws + 2 * szP);
    float* maxbuf = (float*)(ws + 2 * szP + szLg);
    float* sumbuf = (float*)(ws + 2 * szP + szLg + szSeg);
    (void)ws_size; (void)n_in; (void)out_size;

    // 0) init out + segment scratch
    {
        long nOut = (long)N * HD;
        long nSeg = (long)N * HDIM;
        long mx   = nOut > nSeg ? nOut : nSeg;
        int blocks = (int)((mx + 255) / 256);
        init_kernel<<<blocks, 256, 0, stream>>>(out, maxbuf, sumbuf, nOut, nSeg);
    }
    // 1) node projections via WMMA
    {
        int blocks = (N + 127) / 128;
        gemm_proj_kernel<<<blocks, 256, 0, stream>>>(nodes, Ws, bs, Wr, br,
                                                     Ps, Pr, N);
    }
    // 2) logits + segment max
    {
        int blocks = (int)((EH + 255) / 256);
        logits_kernel<<<blocks, 256, 0, stream>>>(Ps, Pr, senders, receivers,
                                                  attn_k, attn_b, logits,
                                                  maxbuf, EH);
    }
    // 3) exp + segment sum
    {
        int blocks = (int)((EH + 255) / 256);
        expsum_kernel<<<blocks, 256, 0, stream>>>(receivers, logits, maxbuf,
                                                  sumbuf, EH);
    }
    // 4) weighted scatter to output nodes
    {
        int blocks = (int)((EH + 255) / 256);
        scatter_kernel<<<blocks, 256, 0, stream>>>(Ps, senders, receivers,
                                                   logits, sumbuf, out, EH);
    }
}
